// SE3TransformerWrapper_4801773437165
// MI455X (gfx1250) — matrix operand order repaired and checked by reading, not compiled, not run on hardware
//
#include <hip/hip_runtime.h>
#include <hip/hip_bf16.h>

typedef __attribute__((ext_vector_type(16))) _Float16 v16h;
typedef __attribute__((ext_vector_type(8)))  float    v8f;
typedef __attribute__((ext_vector_type(4)))  unsigned int v4u;

union HFrag { v4u u[2]; v16h h; };

__device__ inline float sigf(float x) { return 1.0f / (1.0f + expf(-x)); }

__device__ inline float dot16(const float* a, const float* w) {
    float s = 0.f;
#pragma unroll
    for (int d = 0; d < 16; d++) s = fmaf(a[d], w[d], s);
    return s;
}

__device__ inline void ln16p(const float* x, const float* g, const float* b, float* o) {
    float xx[16];
    float m = 0.f;
#pragma unroll
    for (int d = 0; d < 16; d++) { xx[d] = x[d]; m += xx[d]; }
    m *= 0.0625f;
    float v = 0.f;
#pragma unroll
    for (int d = 0; d < 16; d++) { float t = xx[d] - m; v += t * t; }
    v *= 0.0625f;
    float rs = rsqrtf(v + 1e-5f);
#pragma unroll
    for (int d = 0; d < 16; d++) o[d] = (xx[d] - m) * rs * g[d] + b[d];
}

// A fragment: 16x32 f16, row r = lane&15; elems 0-7 = K[k0 + kh*8 ..], elems 8-15 = K[k0+16+kh*8 ..]
__device__ inline v16h loadA(const _Float16* base, int stride, int k0) {
    int lane = threadIdx.x & 31;
    const _Float16* p = base + (size_t)(lane & 15) * stride + k0 + (lane >> 4) * 8;
    HFrag f;
    f.u[0] = *(const v4u*)(p);
    f.u[1] = *(const v4u*)(p + 16);
    return f.h;
}

// B fragment from B^T[N][K] buffer: col n = lane&15; elems 0-7 = K[k0 + kh*8 ..], elems 8-15 = K[k0+16+kh*8 ..]
__device__ inline v16h loadBT(const _Float16* base, int stride, int k0) {
    int lane = threadIdx.x & 31;
    const _Float16* p = base + (size_t)(lane & 15) * stride + k0 + (lane >> 4) * 8;
    HFrag f;
    f.u[0] = *(const v4u*)(p);
    f.u[1] = *(const v4u*)(p + 16);
    return f.h;
}

__device__ inline v8f wmma16(v16h a, v16h b, v8f c) {
    return __builtin_amdgcn_wmma_f32_16x16x32_f16(false, a, false, b, (short)0, c, false, false);
}

// ---------------- precompute kernels ----------------

__global__ void k_prep(const float* hs_rec, const float* hs_lig, const float* labelidx,
                       const float* xyz_lig, const float* WrsW, const float* Wrsb,
                       const float* WlsW, const float* Wlsb,
                       float* h_r, float* h_l, float* x_l) {
    int t = blockIdx.x * blockDim.x + threadIdx.x;
    if (t < 512) {
        float x[16];
#pragma unroll
        for (int d = 0; d < 16; d++) x[d] = hs_rec[t * 16 + d];
#pragma unroll
        for (int d = 0; d < 16; d++) h_r[t * 16 + d] = dot16(x, WrsW + d * 16) + Wrsb[d];
    } else if (t < 640) {
        int j = t - 512;
        float acc[16];
        float xl0 = 0.f, xl1 = 0.f, xl2 = 0.f;
#pragma unroll
        for (int d = 0; d < 16; d++) acc[d] = 0.f;
        for (int jj = 0; jj < 128; jj++) {
            float w = labelidx[j * 128 + jj];
#pragma unroll
            for (int d = 0; d < 16; d++) acc[d] += w * hs_lig[jj * 16 + d];
            xl0 += w * xyz_lig[jj * 3 + 0];
            xl1 += w * xyz_lig[jj * 3 + 1];
            xl2 += w * xyz_lig[jj * 3 + 2];
        }
#pragma unroll
        for (int d = 0; d < 16; d++) h_l[j * 16 + d] = dot16(acc, WlsW + d * 16) + Wlsb[d];
        x_l[j * 3 + 0] = xl0; x_l[j * 3 + 1] = xl1; x_l[j * 3 + 2] = xl2;
    }
}

__global__ void k_zinit(const float* h_r, const float* h_l, float* z) {
    int i = blockIdx.x, j = threadIdx.x;
    size_t ij = (size_t)i * 128 + j;
#pragma unroll
    for (int d = 0; d < 16; d++)
        z[ij * 16 + d] = sigf(h_r[i * 16 + d] * h_l[j * 16 + d]);
}

__global__ void k_bin(const float* xyz, int n, unsigned char* out) {
    int t = blockIdx.x * blockDim.x + threadIdx.x;
    if (t >= n * n) return;
    int i = t / n, k = t % n;
    float d2 = 0.f;
#pragma unroll
    for (int l = 0; l < 3; l++) { float dd = xyz[i * 3 + l] - xyz[k * 3 + l]; d2 += dd * dd; }
    float dis = fminf(sqrtf(fmaxf(d2, 1e-12f)), 15.0f);
    int idx = (int)floorf((dis + 1.0f) * 0.5f);
    idx = max(0, min(15, idx));
    out[t] = (unsigned char)idx;
}

// ---------------- per-stack kernels ----------------

// pp/cp tables: ln(one-hot) is fixed per bin -> gated linear is a [16][128] table
__global__ void k_tables(const float* g1W, const float* g1b, const float* g2W, const float* g2b,
                         const float* l1W, const float* l1b, const float* l2W, const float* l2b,
                         const float* lng, const float* lnb, float* tPP, float* tCP) {
    int t = blockIdx.x * blockDim.x + threadIdx.x;
    if (t >= 2048) return;
    int b = t >> 7, c = t & 127;
    const float mean = 0.0625f;
    const float rs = rsqrtf(15.0f / 256.0f + 1e-5f);
    float u[16];
#pragma unroll
    for (int d = 0; d < 16; d++)
        u[d] = (((d == b) ? 1.0f : 0.0f) - mean) * rs * lng[d] + lnb[d];
    float gp = dot16(u, g2W + c * 16) + g2b[c];
    float lp = dot16(u, l2W + c * 16) + l2b[c];
    tPP[b * 128 + c] = sigf(gp) * lp;
    float gc = dot16(u, g1W + c * 16) + g1b[c];
    float lc = dot16(u, l1W + c * 16) + l1b[c];
    tCP[b * 128 + c] = sigf(gc) * lc;
}

// ab1 (as [c][j_m][k_n]), ab2 (as [c][i_n][k_m]), eg gate
__global__ __launch_bounds__(128) void k_ab(const float* z, const float* lng, const float* lnb,
        const float* g1W, const float* g1b, const float* l1W, const float* l1b,
        const float* g2W, const float* g2b, const float* l2W, const float* l2b,
        const float* egW, const float* egb,
        _Float16* ab1, _Float16* ab2, float* ggate) {
    __shared__ float sW[4][128][16];
    __shared__ float sB[4][128];
    __shared__ float sEg[16][16];
    __shared__ float sEgb[16];
    int i = blockIdx.x, j = threadIdx.x;
    for (int t = j; t < 2048; t += 128) {
        (&sW[0][0][0])[t] = g1W[t];
        (&sW[1][0][0])[t] = l1W[t];
        (&sW[2][0][0])[t] = g2W[t];
        (&sW[3][0][0])[t] = l2W[t];
    }
    sB[0][j] = g1b[j]; sB[1][j] = l1b[j]; sB[2][j] = g2b[j]; sB[3][j] = l2b[j];
    for (int t = j; t < 256; t += 128) (&sEg[0][0])[t] = egW[t];
    if (j < 16) sEgb[j] = egb[j];
    __syncthreads();
    size_t ij = (size_t)i * 128 + j;
    float zl[16];
    ln16p(z + ij * 16, lng, lnb, zl);
    for (int c = 0; c < 128; c++) {
        float a1 = sigf(dot16(zl, sW[0][c]) + sB[0][c]) * (dot16(zl, sW[1][c]) + sB[1][c]);
        float a2 = sigf(dot16(zl, sW[2][c]) + sB[2][c]) * (dot16(zl, sW[3][c]) + sB[3][c]);
        ab1[((size_t)c * 128 + j) * 512 + i] = (_Float16)a1;  // B^T layout for block1
        ab2[((size_t)c * 512 + i) * 128 + j] = (_Float16)a2;  // A layout for block2
    }
#pragma unroll
    for (int d = 0; d < 16; d++)
        ggate[ij * 16 + d] = sigf(dot16(zl, sEg[d]) + sEgb[d]);
}

// cp materialized f16 [c][j][k] (B^T for block2)
__global__ void k_cpmat(const unsigned char* bin_lig, const float* tCP, _Float16* cpmat) {
    int t = blockIdx.x * blockDim.x + threadIdx.x;
    if (t >= 128 * 128 * 128) return;
    int c = t >> 14;
    int rem = t & 16383;
    int j = rem >> 7, k = rem & 127;
    cpmat[t] = (_Float16)tCP[(int)bin_lig[j * 128 + k] * 128 + c];
}

// per (channel c, 16-row i-tile): blocksum[c][i][j] = pp_c @ ab1_c + ab2_c @ cp_c^T
__global__ __launch_bounds__(32) void k_block(const unsigned char* bin_rec, const float* tPP,
        const _Float16* ab1, const _Float16* ab2, const _Float16* cpmat, float* bsum) {
    __shared__ _Float16 Apan[16][512];
    __shared__ float tcol[16];
    int c = blockIdx.x;
    int i0 = blockIdx.y * 16;
    int lane = threadIdx.x;
    if (lane < 16) tcol[lane] = tPP[lane * 128 + c];
    __syncthreads();
    {
        int r = lane & 15, half = lane >> 4;
        const unsigned char* br = bin_rec + (size_t)(i0 + r) * 512 + half * 256;
        _Float16* ap = &Apan[r][half * 256];
        for (int k = 0; k < 256; k++) ap[k] = (_Float16)tcol[br[k]];
    }
    __syncthreads();
    const _Float16* a2base = ab2 + ((size_t)c * 512 + i0) * 128;
    for (int jt = 0; jt < 8; jt++) {
        v8f acc = {0.f, 0.f, 0.f, 0.f, 0.f, 0.f, 0.f, 0.f};
        const _Float16* b1base = ab1 + ((size_t)c * 128 + jt * 16) * 512;
#pragma unroll
        for (int ks = 0; ks < 16; ks++) {          // block1: K = 512
            v16h a = loadA(&Apan[0][0], 512, ks * 32);
            v16h b = loadBT(b1base, 512, ks * 32);
            acc = wmma16(a, b, acc);
        }
        const _Float16* b2base = cpmat + ((size_t)c * 128 + jt * 16) * 128;
#pragma unroll
        for (int ks = 0; ks < 4; ks++) {           // block2: K = 128
            v16h a = loadA(a2base, 128, ks * 32);
            v16h b = loadBT(b2base, 128, ks * 32);
            acc = wmma16(a, b, acc);
        }
        float* out = bsum + ((size_t)c * 512 + i0) * 128 + jt * 16;
        int col = lane & 15, rb = (lane >> 4) * 8;
#pragma unroll
        for (int vi = 0; vi < 8; vi++)
            out[(size_t)(rb + vi) * 128 + col] = acc[vi];
    }
}

// ln over C, las linear, gated residual
__global__ __launch_bounds__(128) void k_triout(const float* bsum, const float* ggate,
        const float* lncg, const float* lncb, const float* lasW, const float* lasb, float* z) {
    __shared__ float sLas[16][128];
    __shared__ float sG[128], sB2[128];
    int i = blockIdx.x, j = threadIdx.x;
    for (int t = j; t < 2048; t += 128) (&sLas[0][0])[t] = lasW[t];
    sG[j] = lncg[j]; sB2[j] = lncb[j];
    __syncthreads();
    const float* bp = bsum + (size_t)i * 128 + j;
    float s = 0.f, ss = 0.f;
    for (int c = 0; c < 128; c++) { float v = bp[(size_t)c * 65536]; s += v; ss += v * v; }
    float mean = s * (1.f / 128.f);
    float var = ss * (1.f / 128.f) - mean * mean;
    float rs = rsqrtf(fmaxf(var, 0.f) + 1e-5f);
    float out[16];
#pragma unroll
    for (int d = 0; d < 16; d++) out[d] = lasb[d];
    for (int c = 0; c < 128; c++) {
        float nb = (bp[(size_t)c * 65536] - mean) * rs * sG[c] + sB2[c];
#pragma unroll
        for (int d = 0; d < 16; d++) out[d] = fmaf(nb, sLas[d][c], out[d]);
    }
    size_t ij = (size_t)i * 128 + j;
#pragma unroll
    for (int d = 0; d < 16; d++) z[ij * 16 + d] += ggate[ij * 16 + d] * out[d];
}

// per-row attention: 4 waves cooperate on each head sequentially
__global__ __launch_bounds__(128) void k_att(const float* z, const float* lng, const float* lnb,
        const float* qW, const float* kW, const float* vW, float* Obuf) {
    __shared__ float zlS[128][16];
    __shared__ _Float16 qS[128][32];
    __shared__ _Float16 kS[128][32];
    __shared__ _Float16 vtS[32][128];
    __shared__ float sS[128][128];
    __shared__ _Float16 wS[128][128];
    int i = blockIdx.x;
    int tid = threadIdx.x;
    int lane = tid & 31, wid = tid >> 5;
    ln16p(z + ((size_t)i * 128 + tid) * 16, lng, lnb, zlS[tid]);
    __syncthreads();
    const float scale = 0.1767766952966369f;   // 32^-0.5
    for (int h = 0; h < 4; h++) {
        for (int idx = tid; idx < 4096; idx += 128) {
            int jj = idx >> 5, t = idx & 31;
            const float* wq = qW + (h * 32 + t) * 16;
            const float* wk = kW + (h * 32 + t) * 16;
            const float* wv = vW + (h * 32 + t) * 16;
            float aq = 0.f, ak = 0.f, av = 0.f;
#pragma unroll
            for (int d = 0; d < 16; d++) {
                float zz = zlS[jj][d];
                aq = fmaf(zz, wq[d], aq); ak = fmaf(zz, wk[d], ak); av = fmaf(zz, wv[d], av);
            }
            qS[jj][t] = (_Float16)(aq * scale);
            kS[jj][t] = (_Float16)ak;
            vtS[t][jj] = (_Float16)av;
        }
        __syncthreads();
        for (int tile = wid; tile < 64; tile += 4) {   // scores q @ k^T
            int mt = tile >> 3, nt = tile & 7;
            v8f acc = {0.f, 0.f, 0.f, 0.f, 0.f, 0.f, 0.f, 0.f};
            v16h a = loadA(&qS[mt * 16][0], 32, 0);
            v16h b = loadBT(&kS[nt * 16][0], 32, 0);
            acc = wmma16(a, b, acc);
            int col = lane & 15, rb = (lane >> 4) * 8;
#pragma unroll
            for (int vi = 0; vi < 8; vi++)
                sS[mt * 16 + rb + vi][nt * 16 + col] = acc[vi];
        }
        __syncthreads();
        {   // softmax over key axis, one query row per thread
            float mx = -1e30f;
            for (int k2 = 0; k2 < 128; k2++) mx = fmaxf(mx, sS[tid][k2]);
            float sum = 0.f;
            for (int k2 = 0; k2 < 128; k2++) { float e = expf(sS[tid][k2] - mx); sS[tid][k2] = e; sum += e; }
            float inv = 1.0f / sum;
            for (int k2 = 0; k2 < 128; k2++) wS[tid][k2] = (_Float16)(sS[tid][k2] * inv);
        }
        __syncthreads();
        for (int tile = wid; tile < 16; tile += 4) {   // O = W @ V
            int mt = tile >> 1, nt = tile & 1;
            v8f acc = {0.f, 0.f, 0.f, 0.f, 0.f, 0.f, 0.f, 0.f};
#pragma unroll
            for (int ks = 0; ks < 4; ks++) {
                v16h a = loadA(&wS[mt * 16][0], 128, ks * 32);
                v16h b = loadBT(&vtS[nt * 16][0], 128, ks * 32);
                acc = wmma16(a, b, acc);
            }
            int col = lane & 15, rb = (lane >> 4) * 8;
            float* op = Obuf + ((size_t)i * 128 + mt * 16) * 128 + h * 32 + nt * 16 + col;
#pragma unroll
            for (int vi = 0; vi < 8; vi++)
                op[(size_t)(rb + vi) * 128] = acc[vi];
        }
        __syncthreads();
    }
}

__global__ __launch_bounds__(128) void k_attout(float* z, const float* Obuf,
        const float* lng, const float* lnb, const float* gW, const float* gb,
        const float* finW, const float* finb) {
    __shared__ float sGW[128][16];
    __shared__ float sFin[16][128];
    __shared__ float sGb[128];
    int i = blockIdx.x, j = threadIdx.x;
    for (int t = j; t < 2048; t += 128) { (&sGW[0][0])[t] = gW[t]; (&sFin[0][0])[t] = finW[t]; }
    sGb[j] = gb[j];
    __syncthreads();
    size_t ij = (size_t)i * 128 + j;
    float zl[16];
    ln16p(z + ij * 16, lng, lnb, zl);
    float out[16];
#pragma unroll
    for (int d = 0; d < 16; d++) out[d] = finb[d];
    const float* op = Obuf + ij * 128;
    for (int c = 0; c < 128; c++) {
        float go = sigf(dot16(zl, sGW[c]) + sGb[c]) * op[c];
#pragma unroll
        for (int d = 0; d < 16; d++) out[d] = fmaf(go, sFin[d][c], out[d]);
    }
#pragma unroll
    for (int d = 0; d < 16; d++) z[ij * 16 + d] += out[d];
}

__global__ __launch_bounds__(128) void k_trans(float* z, const float* lng, const float* lnb,
        const float* l1W, const float* l1b, const float* l2W, const float* l2b) {
    __shared__ float sL1[64][16];
    __shared__ float sL2[16][64];
    __shared__ float sL1b[64], sL2b[16];
    int i = blockIdx.x, j = threadIdx.x;
    for (int t = j; t < 1024; t += 128) { (&sL1[0][0])[t] = l1W[t]; (&sL2[0][0])[t] = l2W[t]; }
    if (j < 64) sL1b[j] = l1b[j];
    if (j < 16) sL2b[j] = l2b[j];
    __syncthreads();
    size_t ij = (size_t)i * 128 + j;
    float zl[16];
    ln16p(z + ij * 16, lng, lnb, zl);
    float out[16];
#pragma unroll
    for (int d = 0; d < 16; d++) out[d] = sL2b[d];
    for (int t = 0; t < 64; t++) {
        float ht = fmaxf(dot16(zl, sL1[t]) + sL1b[t], 0.f);
#pragma unroll
        for (int d = 0; d < 16; d++) out[d] = fmaf(ht, sL2[d][t], out[d]);
    }
#pragma unroll
    for (int d = 0; d < 16; d++) z[ij * 16 + d] = out[d];
}

__global__ void k_zf(const float* z, const float* lastW, const float* lastb, float* outF) {
    int i = blockIdx.x, j = threadIdx.x;
    size_t ij = (size_t)i * 128 + j;
    float x[16];
#pragma unroll
    for (int d = 0; d < 16; d++) x[d] = z[ij * 16 + d];
    outF[384 + ij] = dot16(x, lastW) + lastb[0];
}

__global__ void k_yrec(float* outF, const float* xyz_rec) {
    int k = threadIdx.x;
    float s0 = 0.f, s1 = 0.f, s2 = 0.f;
    for (int i = 0; i < 512; i++) {
        float f = outF[384 + (size_t)i * 128 + k];
        s0 = fmaf(f, xyz_rec[i * 3 + 0], s0);
        s1 = fmaf(f, xyz_rec[i * 3 + 1], s1);
        s2 = fmaf(f, xyz_rec[i * 3 + 2], s2);
    }
    outF[k * 3 + 0] = s0; outF[k * 3 + 1] = s1; outF[k * 3 + 2] = s2;
}

// ---------------- host launcher ----------------

extern "C" void kernel_launch(void* const* d_in, const int* in_sizes, int n_in,
                              void* d_out, int out_size, void* d_ws, size_t ws_size,
                              hipStream_t stream) {
    (void)in_sizes; (void)n_in; (void)out_size; (void)ws_size;
    const float* hs_rec   = (const float*)d_in[0];
    const float* hs_lig   = (const float*)d_in[1];
    const float* xyz_rec  = (const float*)d_in[2];
    const float* xyz_lig  = (const float*)d_in[3];
    const float* labelidx = (const float*)d_in[4];

    // params in JAX pytree order (sorted keys): Wls, Wrs, att[5], last, trans, tri[5]
    int p = 5;
    auto F = [&](void) { return (const float*)d_in[p++]; };
    const float* WlsW = F(); const float* Wlsb = F();
    const float* WrsW = F(); const float* Wrsb = F();
    struct AttP { const float *finW, *finb, *gW, *gb, *kW, *lnb, *lng, *qW, *vW; } A[5];
    for (int s = 0; s < 5; s++) {
        A[s].finW = F(); A[s].finb = F(); A[s].gW = F(); A[s].gb = F();
        A[s].kW = F(); A[s].lnb = F(); A[s].lng = F(); A[s].qW = F(); A[s].vW = F();
    }
    const float* lastW = F(); const float* lastb = F();
    const float* t_l1W = F(); const float* t_l1b = F();
    const float* t_l2W = F(); const float* t_l2b = F();
    const float* t_lnb = F(); const float* t_lng = F();
    struct TriP { const float *egW, *egb, *g1W, *g1b, *g2W, *g2b, *l1W, *l1b,
                              *l2W, *l2b, *lasW, *lasb, *lnb, *lng, *lncb, *lncg; } T[5];
    for (int s = 0; s < 5; s++) {
        T[s].egW = F(); T[s].egb = F(); T[s].g1W = F(); T[s].g1b = F();
        T[s].g2W = F(); T[s].g2b = F(); T[s].l1W = F(); T[s].l1b = F();
        T[s].l2W = F(); T[s].l2b = F(); T[s].lasW = F(); T[s].lasb = F();
        T[s].lnb = F(); T[s].lng = F(); T[s].lncb = F(); T[s].lncg = F();
    }

    char* ws = (char*)d_ws;
    const size_t MB = (size_t)1 << 20;
    float* zbuf  = (float*)(ws + 0);             // [512][128][16] f32, 4 MB
    float* ggate = (float*)(ws + 4 * MB);        // [512][128][16] f32, 4 MB
    float* h_r = (float*)(ws + 8 * MB);
    float* h_l = (float*)(ws + 8 * MB + 64 * 1024);
    float* x_l = (float*)(ws + 8 * MB + 128 * 1024);
    float* tPP = (float*)(ws + 8 * MB + 192 * 1024);
    float* tCP = (float*)(ws + 8 * MB + 256 * 1024);
    unsigned char* bin_rec = (unsigned char*)(ws + 8 * MB + 320 * 1024);
    unsigned char* bin_lig = (unsigned char*)(ws + 8 * MB + 640 * 1024);
    _Float16* ab1   = (_Float16*)(ws + 9 * MB);  // [128][128][512] f16, 16 MB
    _Float16* ab2   = (_Float16*)(ws + 25 * MB); // [128][512][128] f16, 16 MB
    _Float16* cpmat = (_Float16*)(ws + 41 * MB); // [128][128][128] f16, 4 MB
    float* bsum = (float*)(ws + 45 * MB);        // [128][512][128] f32, 32 MB
    float* Obuf = bsum;                          // aliased: used in a later phase
    float* outF = (float*)d_out;

    k_prep<<<5, 128, 0, stream>>>(hs_rec, hs_lig, labelidx, xyz_lig,
                                  WrsW, Wrsb, WlsW, Wlsb, h_r, h_l, x_l);
    k_zinit<<<512, 128, 0, stream>>>(h_r, h_l, zbuf);
    k_bin<<<(512 * 512) / 256, 256, 0, stream>>>(xyz_rec, 512, bin_rec);
    k_bin<<<(128 * 128) / 256, 256, 0, stream>>>(x_l, 128, bin_lig);

    for (int s = 0; s < 5; s++) {
        k_tables<<<8, 256, 0, stream>>>(T[s].g1W, T[s].g1b, T[s].g2W, T[s].g2b,
                                        T[s].l1W, T[s].l1b, T[s].l2W, T[s].l2b,
                                        T[s].lng, T[s].lnb, tPP, tCP);
        k_ab<<<512, 128, 0, stream>>>(zbuf, T[s].lng, T[s].lnb,
                                      T[s].g1W, T[s].g1b, T[s].l1W, T[s].l1b,
                                      T[s].g2W, T[s].g2b, T[s].l2W, T[s].l2b,
                                      T[s].egW, T[s].egb, ab1, ab2, ggate);
        k_cpmat<<<8192, 256, 0, stream>>>(bin_lig, tCP, cpmat);
        k_block<<<dim3(128, 32), 32, 0, stream>>>(bin_rec, tPP, ab1, ab2, cpmat, bsum);
        k_triout<<<512, 128, 0, stream>>>(bsum, ggate, T[s].lncg, T[s].lncb,
                                          T[s].lasW, T[s].lasb, zbuf);
        k_att<<<512, 128, 0, stream>>>(zbuf, A[s].lng, A[s].lnb,
                                       A[s].qW, A[s].kW, A[s].vW, Obuf);
        k_attout<<<512, 128, 0, stream>>>(zbuf, Obuf, A[s].lng, A[s].lnb,
                                          A[s].gW, A[s].gb, A[s].finW, A[s].finb);
        k_trans<<<512, 128, 0, stream>>>(zbuf, t_lng, t_lnb, t_l1W, t_l1b, t_l2W, t_l2b);
    }
    k_zf<<<512, 128, 0, stream>>>(zbuf, lastW, lastb, outF);
    k_yrec<<<1, 128, 0, stream>>>(outF, xyz_rec);
}
